// KAN_RNN_Layer_56762287784114
// MI455X (gfx1250) — matrix operand order repaired and checked by reading, compiled, not run on hardware
//
#include <hip/hip_runtime.h>
#include <hip/hip_bf16.h>
#include <math.h>

typedef __attribute__((ext_vector_type(16))) _Float16 v16h;
typedef __attribute__((ext_vector_type(8)))  _Float16 v8h;
typedef __attribute__((ext_vector_type(8)))  float    v8f;

#define N_AGENTS 8
#define IN_DIM   2
#define HIDDEN   32
#define M_NETS   16          // N_AGENTS * IN_DIM
#define GATES    96          // 3 * HIDDEN
#define T_STEPS  16
#define B_TOT    1024
#define CELLS    128         // M_NETS * N_AGENTS
#define CELL_H   (16*HIDDEN)     // 512 halves per h tile (16 rows x 32)
#define CELL_W   (GATES*HIDDEN)  // 3072 halves per weight matrix [96][32]
#define WP_LIN_OFF (3*CELLS*CELL_W)            // 1179648 halves
#define WLIN_ELEMS (M_NETS*N_AGENTS*HIDDEN)    // 4096

// LDS layout (bytes):
//   h state   : 2 layers * 128 cells * 512 halves * 2B = 262144
//   x tile    : 16 rows * 16 floats * 4B               =   1024
//   W_lin f16 : 4096 halves * 2B                       =   8192
#define SMEM_H_BYTES  (2*CELLS*CELL_H*2)
#define SMEM_X_BYTES  (16*16*4)
#define SMEM_WL_BYTES (WLIN_ELEMS*2)
#define SMEM_TOTAL    (SMEM_H_BYTES + SMEM_X_BYTES + SMEM_WL_BYTES)

__device__ __forceinline__ v8f wmma16x16x32(v16h a, v16h b, v8f c) {
  // D = A(16x32 f16) * B(32x16 f16) + C(16x16 f32)
  return __builtin_amdgcn_wmma_f32_16x16x32_f16(false, a, false, b, (short)0, c,
                                                false, false);
}

// Branch-free fast activations: v_exp_f32 + v_rcp_f32, no IEEE-div sequence,
// no EXEC-divergent range-check branches (WMMA requires EXEC all-ones).
__device__ __forceinline__ float sigm_fast(float x) {
  return __builtin_amdgcn_rcpf(1.0f + __expf(-x));
}
__device__ __forceinline__ float tanh_fast(float x) {
  // tanh(x) = 1 - 2/(exp(2x)+1)
  return __builtin_fmaf(-2.0f, __builtin_amdgcn_rcpf(__expf(2.0f*x) + 1.0f),
                        1.0f);
}

// Load h tile [16 rows][32 k] (f16, row-major in LDS) into the WMMA 16-bit
// A operand in this code's contiguous k order (B uses the same): lane holds row = lane&15;
//   halves j=0..15 -> K = (lane>>4)*16 + j
__device__ __forceinline__ v16h load_A_lds(const _Float16* base, int lane) {
  const int row = lane & 15, hi = lane >> 4;
  const _Float16* p = base + row*HIDDEN + hi*16;
  v8h lo = *(const v8h*)(p);        // K = hi*16 .. hi*16+7
  v8h hh = *(const v8h*)(p + 8);    // K = hi*16+8 .. hi*16+15
  v16h out;
#pragma unroll
  for (int j = 0; j < 8; ++j) { out[j] = lo[j]; out[j+8] = hh[j]; }
  return out;
}

// One GRU cell update for one 16-row batch tile.
// LAYER 0: gi from x (D=2, VALU). LAYER 1: gi = WMMA(h0_new, W_ih1).
// All WMMA accumulators start from literal 0 (inline SRC2=0, no v_movs);
// biases are applied in the post-WMMA elementwise math.
template<int LAYER>
__device__ __forceinline__ void cell_step(_Float16* h_lds, const float* x_lds,
                                          const _Float16* wp,
                                          const float* Wih0, const float* bih,
                                          const float* bhh, int cell, int lane) {
  const int n = cell & 7;
  _Float16* hcur = h_lds + (size_t)(LAYER*CELLS + cell)*CELL_H;
  const v16h a_h = load_A_lds(hcur, lane);
  const v16h a_in = (LAYER == 1) ? load_A_lds(h_lds + (size_t)cell*CELL_H, lane)
                                 : a_h;
  const _Float16* whh  = wp + (size_t)(LAYER*CELLS + cell)*CELL_W;
  const _Float16* wih1 = wp + (size_t)(2*CELLS   + cell)*CELL_W;
  const float* bihc  = bih  + LAYER*(CELLS*GATES) + cell*GATES;
  const float* bhhc  = bhh  + LAYER*(CELLS*GATES) + cell*GATES;
  const float* wih0c = Wih0 + cell*(GATES*IN_DIM);
  const int col = lane & 15, hi = lane >> 4;

  // Preload x rows for layer-0 gi (row = v + hi*8)
  float xv0[8], xv1[8];
  if (LAYER == 0) {
#pragma unroll
    for (int v = 0; v < 8; ++v) {
      const int row = v + hi*8;
      xv0[v] = x_lds[row*16 + n*2 + 0];
      xv1[v] = x_lds[row*16 + n*2 + 1];
    }
  }

  float rr[2][8], zz[2][8], nn[2][8];
#pragma unroll
  for (int blk = 0; blk < 6; ++blk) {
    const int g = blk*16 + col;   // gate index handled by this lane's column
    const float bi = bihc[g];
    const float bh = bhhc[g];

    // gh_w = h * W_hh^T (no bias; inline-zero accumulator)
    const v16h bmat = *(const v16h*)(whh + (size_t)g*HIDDEN + hi*16);
    v8f cgh = {};
    cgh = wmma16x16x32(a_h, bmat, cgh);

    // gi_w (no bias)
    v8f giw;
    if (LAYER == 0) {
      const float w0 = wih0c[g*2 + 0], w1 = wih0c[g*2 + 1];
#pragma unroll
      for (int v = 0; v < 8; ++v) giw[v] = xv0[v]*w0 + xv1[v]*w1;
    } else {
      const v16h bmat2 = *(const v16h*)(wih1 + (size_t)g*HIDDEN + hi*16);
      v8f cgi = {};
      giw = wmma16x16x32(a_in, bmat2, cgi);
    }

    if (blk < 2) {                     // r gates (g in [0,32))
      const float bs = bi + bh;
#pragma unroll
      for (int v = 0; v < 8; ++v) rr[blk][v] = sigm_fast(giw[v] + cgh[v] + bs);
    } else if (blk < 4) {              // z gates (g in [32,64))
      const float bs = bi + bh;
#pragma unroll
      for (int v = 0; v < 8; ++v) zz[blk-2][v] = sigm_fast(giw[v] + cgh[v] + bs);
    } else {                           // n gates: tanh((gi+bi) + r*(gh+bh))
#pragma unroll
      for (int v = 0; v < 8; ++v) {
        const float ghn = cgh[v] + bh;
        const float gin = giw[v] + bi;
        nn[blk-4][v] = tanh_fast(__builtin_fmaf(rr[blk-4][v], ghn, gin));
      }
    }
  }

  // h' = (1-z)*n + z*h ; write back as f16 (C layout: row=v+8*hi, col-block q)
#pragma unroll
  for (int q = 0; q < 2; ++q) {
#pragma unroll
    for (int v = 0; v < 8; ++v) {
      const int row = v + hi*8;
      const int hd  = q*16 + col;
      const float hold = (float)hcur[row*HIDDEN + hd];
      const float z    = zz[q][v];
      const float hnew = __builtin_fmaf(z, hold - nn[q][v], nn[q][v]);
      hcur[row*HIDDEN + hd] = (_Float16)hnew;
    }
  }
}

// Convert W_hh (both layers), W_ih1 and W_lin to f16 into workspace.
// Layout (halves): [hh L0: 128 cells][hh L1: 128 cells][ih1: 128 cells]
// (each cell [96 gates][32 k] contiguous), then [W_lin: 4096].
__global__ void prepack_f16(const float* __restrict__ Whh,
                            const float* __restrict__ Wih1,
                            const float* __restrict__ Wlin,
                            _Float16* __restrict__ wp) {
  const int i = blockIdx.x * 256 + threadIdx.x;
  const int nhh  = 2*CELLS*CELL_W;          // 786432
  const int nih1 = nhh + CELLS*CELL_W;      // 1179648 == WP_LIN_OFF
  const int ntot = nih1 + WLIN_ELEMS;       // 1183744
  if (i < nhh)       wp[i] = (_Float16)Whh[i];
  else if (i < nih1) wp[i] = (_Float16)Wih1[i - nhh];
  else if (i < ntot) wp[i] = (_Float16)Wlin[i - nih1];
}

__global__ __launch_bounds__(256)
void kan_rnn_fused(const float* __restrict__ x_in,
                   const float* __restrict__ Wih0,
                   const float* __restrict__ bih,
                   const float* __restrict__ bhh,
                   const float* __restrict__ blin,
                   const _Float16* __restrict__ wp,
                   float* __restrict__ outs,
                   float* __restrict__ ctrl) {
  extern __shared__ char smem[];
  _Float16* h_lds  = (_Float16*)smem;
  float*    x_lds  = (float*)(smem + SMEM_H_BYTES);
  _Float16* wl_lds = (_Float16*)(smem + SMEM_H_BYTES + SMEM_X_BYTES);

  const int tid  = threadIdx.x;
  const int lane = tid & 31;
  const int wave = tid >> 5;
  const int row0 = blockIdx.x * 16;   // 16 batch rows per workgroup

  // zero h state (f16), 262144B = 16384 float4
  float4* hz = (float4*)h_lds;
  const float4 z4 = {0.f, 0.f, 0.f, 0.f};
  for (int i = tid; i < SMEM_H_BYTES/16; i += 256) hz[i] = z4;
  // stage x tile: rows row0..row0+15, 16 floats (N*D) per row
  x_lds[tid] = x_in[row0*16 + tid];
  // stage W_lin (4096 f16)
  for (int i = tid; i < WLIN_ELEMS; i += 256) wl_lds[i] = wp[WP_LIN_OFF + i];
  __syncthreads();

  for (int t = 0; t < T_STEPS; ++t) {
    // ---- GRU layers: each wave owns 16 cells, both layers fused per cell ----
    for (int ci = 0; ci < 16; ++ci) {
      const int cell = ci*8 + wave;
      cell_step<0>(h_lds, x_lds, wp, Wih0, bih, bhh, cell, lane);
      cell_step<1>(h_lds, x_lds, wp, Wih0, bih, bhh, cell, lane);
    }
    __syncthreads();

    // ---- output head: thread = (m, row); packed f16 relu-dot ----
    const int m   = tid >> 4;
    const int row = tid & 15;
    const _Float16* h1m = h_lds + (size_t)(CELLS + m*N_AGENTS)*CELL_H;
    const v8h z8 = {};
    v8h acc8 = {};
#pragma unroll
    for (int n = 0; n < N_AGENTS; ++n) {
      const v8h* hc = (const v8h*)(h1m + (size_t)n*CELL_H + row*HIDDEN);
      const v8h* wl = (const v8h*)(wl_lds + m*(N_AGENTS*HIDDEN) + n*HIDDEN);
#pragma unroll
      for (int k8 = 0; k8 < HIDDEN/8; ++k8) {
        const v8h r = __builtin_elementwise_max(hc[k8], z8);  // v_pk_max
        acc8 = r * wl[k8] + acc8;                             // v_pk_fma
      }
    }
    float acc = blin[m];
#pragma unroll
    for (int j = 0; j < 8; ++j) acc += (float)acc8[j];

    const float u = fminf(fmaxf(acc, -5.0f), 5.0f);
    const int ag = m >> 1, di = m & 1;          // m = agent*2 + dim
    const float x_old = x_lds[row*16 + ag*2 + di];
    const int gidx = (t*B_TOT + (row0 + row))*(N_AGENTS*IN_DIM) + ag*2 + di;
    outs[gidx] = x_old;                          // scan emits pre-step x
    ctrl[gidx] = u;
    float x_new = __builtin_fmaf(0.1f, u, x_old);  // LAM = 0.1, L[i] = 1.0
    x_new = fminf(fmaxf(x_new, 0.0f), 1.0f);
    x_lds[row*16 + ag*2 + di] = x_new;
    __syncthreads();
  }
}

extern "C" void kernel_launch(void* const* d_in, const int* in_sizes, int n_in,
                              void* d_out, int out_size, void* d_ws, size_t ws_size,
                              hipStream_t stream) {
  const float* x    = (const float*)d_in[0];  // [B, N, D]
  const float* Wih0 = (const float*)d_in[1];  // [M, N, G, D]
  const float* Wih1 = (const float*)d_in[2];  // [M, N, G, H]
  const float* Whh  = (const float*)d_in[3];  // [2, M, N, G, H]
  const float* bih  = (const float*)d_in[4];  // [2, M, N, G]
  const float* bhh  = (const float*)d_in[5];  // [2, M, N, G]
  const float* Wlin = (const float*)d_in[6];  // [M, N*H]
  const float* blin = (const float*)d_in[7];  // [M]
  _Float16* wp = (_Float16*)d_ws;             // 1183744 halves = 2.26 MB
  float* outs = (float*)d_out;
  float* ctrl = outs + (size_t)T_STEPS*B_TOT*N_AGENTS*IN_DIM;

  // 1) weights -> f16 packed for WMMA B-operands + packed-f16 output head
  {
    const int ntot = WP_LIN_OFF + WLIN_ELEMS;   // 1183744
    prepack_f16<<<(ntot + 255)/256, 256, 0, stream>>>(Whh, Wih1, Wlin, wp);
  }

  // 2) fused recurrence: 64 workgroups (one per 16-row batch tile), 8 waves
  (void)hipFuncSetAttribute((const void*)kan_rnn_fused,
                            hipFuncAttributeMaxDynamicSharedMemorySize,
                            (int)SMEM_TOTAL);
  kan_rnn_fused<<<B_TOT/16, 256, SMEM_TOTAL, stream>>>(
      x, Wih0, bih, bhh, blin, wp, outs, ctrl);
}
